// MultiHeadAttentionLayer_77876347011464
// MI455X (gfx1250) — matrix operand order repaired and checked by reading, not compiled, not run on hardware
//
#include <hip/hip_runtime.h>
#include <hip/hip_bf16.h>

// ---------------------------------------------------------------------------
// MI455X (gfx1250) fused multi-head attention layer.
// B=4, S=2048, EMB=1024, H=16, DK=DV=64.
// - All matmuls on v_wmma_f32_16x16x32_bf16 (f32 accumulate, bf16 operands).
// - Tile staging via GLOBAL_LOAD_ASYNC_TO_LDS_B128 (ASYNCcnt) w/ double buffer.
// - 1 GiB attention output streamed LDS->HBM with
//   GLOBAL_STORE_ASYNC_FROM_LDS_B128, overlapped with the ctx GEMM.
// ---------------------------------------------------------------------------

typedef __attribute__((ext_vector_type(16))) __bf16 v16bf;
typedef __attribute__((ext_vector_type(8)))  __bf16 v8bf;
typedef __attribute__((ext_vector_type(8)))  float  v8f;
typedef int v4i __attribute__((vector_size(16)));  // async builtin ptr type

#define B_   4
#define S_   2048
#define EMB_ 1024
#define H_   16
#define M_   (B_ * S_)   // 8192

#if __has_builtin(__builtin_amdgcn_global_load_async_to_lds_b128) && \
    __has_builtin(__builtin_amdgcn_s_wait_asynccnt)
#define USE_ASYNC 1
#else
#define USE_ASYNC 0
#endif
#if __has_builtin(__builtin_amdgcn_global_store_async_from_lds_b128) && \
    __has_builtin(__builtin_amdgcn_s_wait_asynccnt)
#define USE_ASYNC_ST 1
#else
#define USE_ASYNC_ST 0
#endif

// 16-byte global -> LDS copy (async DMA when available)
__device__ __forceinline__ void async_cp_b128(const void* g, void* l) {
#if USE_ASYNC
  __builtin_amdgcn_global_load_async_to_lds_b128((v4i*)g, (v4i*)l, 0, 0);
#else
  *(float4*)l = *(const float4*)g;
#endif
}

// 16-byte LDS -> global copy (async DMA when available)
__device__ __forceinline__ void async_st_b128(void* g, const void* l) {
#if USE_ASYNC_ST
  __builtin_amdgcn_global_store_async_from_lds_b128((v4i*)g, (v4i*)l, 0, 0);
#else
  *(float4*)g = *(const float4*)l;
#endif
}

__device__ __forceinline__ void wait_async0() {
#if USE_ASYNC || USE_ASYNC_ST
  __builtin_amdgcn_s_wait_asynccnt(0);
#endif
}

__device__ __forceinline__ v8f wmma_bf16(v16bf a, v16bf b, v8f c) {
  return __builtin_amdgcn_wmma_f32_16x16x32_bf16(false, a, false, b, (short)0,
                                                 c, false, false);
}

// A fragment (16x32, M x K). Lane L<16 -> row L, K {0..7,16..23};
// lane L>=16 -> row L-16, K {8..15,24..31}.  (ISA 7.12.2)
__device__ __forceinline__ v16bf load_a_frag(const __bf16* base, int ld) {
  const int lane = threadIdx.x & 31;
  const __bf16* p = base + (lane & 15) * ld + ((lane >> 4) << 3);
  v8bf lo = *(const v8bf*)(p);
  v8bf hi = *(const v8bf*)(p + 16);
  return __builtin_shufflevector(lo, hi, 0, 1, 2, 3, 4, 5, 6, 7,
                                 8, 9, 10, 11, 12, 13, 14, 15);
}

// B fragment (32x16, K x N) from row-major N x K buffer; lane L<16 -> col L,
// K {0..7,16..23}; lane L>=16 -> col L-16, K {8..15,24..31}.  (matches A)
__device__ __forceinline__ v16bf load_b_frag(const __bf16* base, int ld) {
  const int lane = threadIdx.x & 31;
  const __bf16* p = base + (lane & 15) * ld + ((lane >> 4) << 3);
  v8bf lo = *(const v8bf*)(p);
  v8bf hi = *(const v8bf*)(p + 16);
  return __builtin_shufflevector(lo, hi, 0, 1, 2, 3, 4, 5, 6, 7,
                                 8, 9, 10, 11, 12, 13, 14, 15);
}

// C/D element e of lane L -> row (e + 8*(L>>4)), col (L&15).
__device__ __forceinline__ int c_row(int e) {
  return e + (((threadIdx.x & 31) >> 4) << 3);
}
__device__ __forceinline__ int c_col() { return threadIdx.x & 15; }

// ---------------------------------------------------------------------------
__global__ void f32_to_bf16_kernel(const float* __restrict__ src,
                                   __bf16* __restrict__ dst, long n) {
  long i = (long)blockIdx.x * blockDim.x + threadIdx.x;
  long stride = (long)gridDim.x * blockDim.x;
  for (; i < n; i += stride) dst[i] = (__bf16)src[i];
}

// ---------------------------------------------------------------------------
// Projection GEMM: D[m,n] = sum_k X[m,k] * W[n,k] + bias[n]
// M=8192, N=1024, K=1024.  Block tile 64x128, 8 waves (2x4), each wave 32x32.
// Double-buffered LDS, async tile prefetch, 1 barrier per K-step.
// Epilogue scatter:
//   mode 0: dst[((b*H+h)*S + s)*64 + d]   (Q,K as [B,H,S,D])
//   mode 1: dst[((b*H+h)*64 + d)*S + s]   (V transposed [B,H,D,S])
// ---------------------------------------------------------------------------
__global__ __launch_bounds__(256) void gemm_proj_kernel(
    const __bf16* __restrict__ X, const __bf16* __restrict__ W,
    const float* __restrict__ bias, __bf16* __restrict__ dst, int mode) {
  __shared__ __bf16 sA[2][64 * 32];
  __shared__ __bf16 sB[2][128 * 32];
  const int tid = threadIdx.x;
  const int lane = tid & 31;
  const int wave = tid >> 5;
  const int m0 = blockIdx.y * 64;
  const int n0 = blockIdx.x * 128;
  const int wm = (wave & 1) * 32;
  const int wn = (wave >> 1) * 32;

  auto load_tiles = [&](int buf, int kk) {
    {  // A tile 64x32: one b128 per thread
      int idx = tid * 8;
      int r = idx >> 5, c = idx & 31;
      async_cp_b128(X + (long)(m0 + r) * EMB_ + kk + c, &sA[buf][idx]);
    }
    {  // B tile 128x32: two b128 per thread
      int idx = tid * 16;
      int r = idx >> 5, c = idx & 31;
      const __bf16* g = W + (long)(n0 + r) * EMB_ + kk + c;
      async_cp_b128(g, &sB[buf][idx]);
      async_cp_b128(g + 8, &sB[buf][idx + 8]);
    }
  };

  v8f acc[2][2] = {};
  load_tiles(0, 0);
  for (int kk = 0; kk < EMB_; kk += 32) {
    const int cur = (kk >> 5) & 1;
    wait_async0();       // this wave's DMA into buf[cur] done
    __syncthreads();     // everyone's DMA done & prior reads of buf[cur^1] done
    if (kk + 32 < EMB_) load_tiles(cur ^ 1, kk + 32);

    v16bf a0 = load_a_frag(&sA[cur][wm * 32], 32);
    v16bf a1 = load_a_frag(&sA[cur][(wm + 16) * 32], 32);
    v16bf b0 = load_b_frag(&sB[cur][wn * 32], 32);
    v16bf b1 = load_b_frag(&sB[cur][(wn + 16) * 32], 32);
    acc[0][0] = wmma_bf16(a0, b0, acc[0][0]);
    acc[0][1] = wmma_bf16(a0, b1, acc[0][1]);
    acc[1][0] = wmma_bf16(a1, b0, acc[1][0]);
    acc[1][1] = wmma_bf16(a1, b1, acc[1][1]);
  }

#pragma unroll
  for (int i = 0; i < 2; i++)
#pragma unroll
    for (int j = 0; j < 2; j++)
#pragma unroll
      for (int e = 0; e < 8; e++) {
        int m = m0 + wm + i * 16 + c_row(e);
        int n = n0 + wn + j * 16 + c_col();
        float v = acc[i][j][e] + bias[n];
        int b = m >> 11, s = m & (S_ - 1);
        int h = n >> 6, d = n & 63;
        long o = (mode == 0)
                     ? ((((long)b * H_ + h) * S_ + s) * 64 + d)
                     : ((((long)b * H_ + h) * 64 + d) * S_ + s);
        dst[o] = (__bf16)v;
      }
}

// ---------------------------------------------------------------------------
// Fused attention: one block = 16 query rows of one (b,h).
//  P1: scores[16][2048] -> LDS via WMMA (K streamed from L2)
//  P2: softmax in LDS; async-stream f32 attn strip LDS->HBM (overlaps P3);
//      keep bf16 copy in LDS
//  P3: ctx = attn @ V from the LDS bf16 copy (V pre-transposed, L2-hot)
// ---------------------------------------------------------------------------
#define LDW 2056  // padded row (16B-aligned, breaks bank conflicts)

__global__ __launch_bounds__(256) void attn_fused_kernel(
    const __bf16* __restrict__ q, const __bf16* __restrict__ k,
    const __bf16* __restrict__ vt, const unsigned char* __restrict__ mask,
    float* __restrict__ attn_out, __bf16* __restrict__ ctx) {
  extern __shared__ char smem[];
  float* s_s = (float*)smem;                                    // 16*LDW f32
  __bf16* s_a = (__bf16*)(smem + 16 * LDW * 4);                 // 16*LDW bf16
  __bf16* s_q = (__bf16*)(smem + 16 * LDW * 4 + 16 * LDW * 2);  // 16*64 bf16
  float* s_red = (float*)(smem + 16 * LDW * 4 + 16 * LDW * 2 + 2048);  // 1024

  const int tid = threadIdx.x, lane = tid & 31, wave = tid >> 5;
  const int bh = blockIdx.y;  // b*H + h
  const int b = bh >> 4, h = bh & 15;
  const int q0 = blockIdx.x * 16;

  // Q tile (1024 contiguous bf16) -> LDS via async DMA
  const __bf16* qbase = q + ((long)bh * S_ + q0) * 64;
  if (tid < 128) async_cp_b128(qbase + tid * 8, s_q + tid * 8);
  wait_async0();
  __syncthreads();

  // ---- P1: scores ---------------------------------------------------------
  v16bf a0 = load_a_frag(s_q, 64);
  v16bf a1 = load_a_frag(s_q + 32, 64);
  const __bf16* kbase = k + (long)bh * S_ * 64;
  for (int jt = wave; jt < S_ / 16; jt += 8) {
    const __bf16* kb = kbase + jt * 16 * 64;
    v16bf b0 = load_b_frag(kb, 64);
    v16bf b1 = load_b_frag(kb + 32, 64);
    v8f c = {};
    c = wmma_bf16(a0, b0, c);
    c = wmma_bf16(a1, b1, c);
    int n = jt * 16 + c_col();
#pragma unroll
    for (int e = 0; e < 8; e++) {
      int m = c_row(e);
      float v = c[e] * 0.125f;  // 1/sqrt(DK)
      if (mask[((long)b * S_ + (q0 + m)) * S_ + n]) v = -__builtin_inff();
      s_s[m * LDW + n] = v;
    }
  }
  __syncthreads();

  // ---- P2: softmax (2 rows per wave), normalized probs kept in LDS --------
  for (int rr = 0; rr < 2; rr++) {
    int r = wave * 2 + rr;
    float* row = s_s + r * LDW;
    float mx = -__builtin_inff();
    for (int c = lane; c < S_; c += 32) mx = fmaxf(mx, row[c]);
#pragma unroll
    for (int o = 16; o >= 1; o >>= 1) mx = fmaxf(mx, __shfl_xor(mx, o, 32));
    float sum = 0.f;
    for (int c = lane; c < S_; c += 32) {
      float e = __expf(row[c] - mx);
      row[c] = e;
      sum += e;
    }
#pragma unroll
    for (int o = 16; o >= 1; o >>= 1) sum += __shfl_xor(sum, o, 32);
    float inv = 1.0f / sum;
    __bf16* arow = s_a + r * LDW;
    for (int c = lane; c < S_; c += 32) {
      float p = row[c] * inv;
      row[c] = p;            // f32 prob, streamed to HBM below
      arow[c] = (__bf16)p;   // bf16 prob for the ctx WMMAs
    }
  }
  __syncthreads();  // all probs in LDS, visible to every wave / the DMA

  // Stream the 16x2048 f32 attn strip LDS -> HBM (attn layout [h*B+b, q, k]).
  // Runs concurrently with the ctx GEMM below; S_ENDPGM drains ASYNCcnt.
  {
    float* obase = attn_out + (((long)h * B_ + b) * S_ + q0) * S_;
#pragma unroll 4
    for (int i = 0; i < 32; i++) {
      int id = tid + i * 256;     // 8192 chunks of 16B
      int r = id >> 9;            // row 0..15
      int c4 = (id & 511) << 2;   // f32 col, multiple of 4
      async_st_b128(obase + (long)r * S_ + c4, s_s + r * LDW + c4);
    }
  }

  // ---- P3: ctx = attn @ V (V pre-transposed [B,H,D,S]) --------------------
  const int nt = wave & 3;      // 16-wide d-subtile
  const int khalf = wave >> 2;  // key-range half
  const __bf16* vb = vt + ((long)bh * 64 + nt * 16) * S_ + khalf * 1024;
  v8f acc = {};
  for (int kc = 0; kc < 1024; kc += 32) {
    v16bf a = load_a_frag(s_a + khalf * 1024 + kc, LDW);
    v16bf bb = load_b_frag(vb + kc, S_);
    acc = wmma_bf16(a, bb, acc);
  }
  if (khalf == 1) {
#pragma unroll
    for (int e = 0; e < 8; e++) s_red[nt * 256 + e * 32 + lane] = acc[e];
  }
  __syncthreads();
  if (khalf == 0) {
#pragma unroll
    for (int e = 0; e < 8; e++) {
      acc[e] += s_red[nt * 256 + e * 32 + lane];
      int m = c_row(e), d = nt * 16 + c_col();
      ctx[((long)b * S_ + (q0 + m)) * EMB_ + h * 64 + d] = (__bf16)acc[e];
    }
  }
}

// ---------------------------------------------------------------------------
// Output projection + bias + residual + LayerNorm, fused per 16-row strip.
// ---------------------------------------------------------------------------
__global__ __launch_bounds__(256) void fc_ln_kernel(
    const __bf16* __restrict__ ctx, const __bf16* __restrict__ W,
    const float* __restrict__ bias, const float* __restrict__ resid,
    const float* __restrict__ g, const float* __restrict__ be,
    float* __restrict__ out) {
  extern __shared__ char smem[];
  __bf16* s_c = (__bf16*)smem;          // 16*1024 bf16 (32 KB)
  float* s_o = (float*)(smem + 32768);  // 16*1024 f32  (64 KB)
  const int tid = threadIdx.x, lane = tid & 31, wave = tid >> 5;
  const long m0 = (long)blockIdx.x * 16;

  const __bf16* cb = ctx + m0 * EMB_;
#pragma unroll
  for (int i = 0; i < 8; i++)
    async_cp_b128(cb + i * 2048 + tid * 8, s_c + i * 2048 + tid * 8);
  wait_async0();
  __syncthreads();

  for (int nt = wave; nt < EMB_ / 16; nt += 8) {
    v8f acc = {};
    const __bf16* wb = W + (long)nt * 16 * EMB_;
    for (int kc = 0; kc < EMB_; kc += 32) {
      v16bf a = load_a_frag(s_c + kc, EMB_);
      v16bf bb = load_b_frag(wb + kc, EMB_);
      acc = wmma_bf16(a, bb, acc);
    }
#pragma unroll
    for (int e = 0; e < 8; e++)
      s_o[c_row(e) * EMB_ + nt * 16 + c_col()] = acc[e];
  }
  __syncthreads();

  for (int rr = 0; rr < 2; rr++) {
    int r = wave * 2 + rr;
    long m = m0 + r;
    float* row = s_o + r * EMB_;
    const float* res = resid + m * EMB_;
    float sum = 0.f, sq = 0.f;
    for (int c = lane; c < EMB_; c += 32) {
      float v = row[c] + bias[c] + res[c];
      row[c] = v;
      sum += v;
      sq += v * v;
    }
#pragma unroll
    for (int o = 16; o >= 1; o >>= 1) {
      sum += __shfl_xor(sum, o, 32);
      sq += __shfl_xor(sq, o, 32);
    }
    float mu = sum * (1.0f / EMB_);
    float var = sq * (1.0f / EMB_) - mu * mu;
    float rs = rsqrtf(var + 1e-5f);
    float* orow = out + m * EMB_;
    for (int c = lane; c < EMB_; c += 32)
      orow[c] = (row[c] - mu) * rs * g[c] + be[c];
  }
}

// ---------------------------------------------------------------------------
extern "C" void kernel_launch(void* const* d_in, const int* in_sizes, int n_in,
                              void* d_out, int out_size, void* d_ws,
                              size_t ws_size, hipStream_t stream) {
  (void)in_sizes; (void)n_in; (void)out_size; (void)ws_size;
  const float* in_q = (const float*)d_in[0];
  const float* in_k = (const float*)d_in[1];
  const float* in_v = (const float*)d_in[2];
  const unsigned char* mask = (const unsigned char*)d_in[3];
  const float* wq_w = (const float*)d_in[4];
  const float* wq_b = (const float*)d_in[5];
  const float* wk_w = (const float*)d_in[6];
  const float* wk_b = (const float*)d_in[7];
  const float* wv_w = (const float*)d_in[8];
  const float* wv_b = (const float*)d_in[9];
  const float* fc_w = (const float*)d_in[10];
  const float* fc_b = (const float*)d_in[11];
  const float* ln_g = (const float*)d_in[12];
  const float* ln_b = (const float*)d_in[13];

  float* outp = (float*)d_out;              // [B,S,EMB]
  float* attnp = outp + (size_t)M_ * EMB_;  // [H*B,S,S]

  char* ws = (char*)d_ws;
  size_t off = 0;
  auto alloc = [&](size_t bytes) -> char* {
    char* p = ws + off;
    off += (bytes + 255) & ~(size_t)255;
    return p;
  };
  const size_t MK = (size_t)M_ * EMB_;
  __bf16* xq = (__bf16*)alloc(MK * 2);
  __bf16* xk = (__bf16*)alloc(MK * 2);
  __bf16* xv = (__bf16*)alloc(MK * 2);
  __bf16* wqb = (__bf16*)alloc((size_t)EMB_ * EMB_ * 2);
  __bf16* wkb = (__bf16*)alloc((size_t)EMB_ * EMB_ * 2);
  __bf16* wvb = (__bf16*)alloc((size_t)EMB_ * EMB_ * 2);
  __bf16* wfb = (__bf16*)alloc((size_t)EMB_ * EMB_ * 2);
  __bf16* qh = (__bf16*)alloc(MK * 2);  // [B,H,S,DK]
  __bf16* kh = (__bf16*)alloc(MK * 2);  // [B,H,S,DK]
  __bf16* vt = (__bf16*)alloc(MK * 2);  // [B,H,DV,S]
  __bf16* cx = (__bf16*)alloc(MK * 2);  // [B,S,H*DV]

  f32_to_bf16_kernel<<<1024, 256, 0, stream>>>(in_q, xq, (long)MK);
  f32_to_bf16_kernel<<<1024, 256, 0, stream>>>(in_k, xk, (long)MK);
  f32_to_bf16_kernel<<<1024, 256, 0, stream>>>(in_v, xv, (long)MK);
  f32_to_bf16_kernel<<<256, 256, 0, stream>>>(wq_w, wqb, (long)EMB_ * EMB_);
  f32_to_bf16_kernel<<<256, 256, 0, stream>>>(wk_w, wkb, (long)EMB_ * EMB_);
  f32_to_bf16_kernel<<<256, 256, 0, stream>>>(wv_w, wvb, (long)EMB_ * EMB_);
  f32_to_bf16_kernel<<<256, 256, 0, stream>>>(fc_w, wfb, (long)EMB_ * EMB_);

  dim3 pgrid(EMB_ / 128, M_ / 64);
  gemm_proj_kernel<<<pgrid, 256, 0, stream>>>(xq, wqb, wq_b, qh, 0);
  gemm_proj_kernel<<<pgrid, 256, 0, stream>>>(xk, wkb, wk_b, kh, 0);
  gemm_proj_kernel<<<pgrid, 256, 0, stream>>>(xv, wvb, wv_b, vt, 1);

  constexpr int ATTN_LDS = 16 * LDW * 4 + 16 * LDW * 2 + 2048 + 4096;
  (void)hipFuncSetAttribute((const void*)attn_fused_kernel,
                            hipFuncAttributeMaxDynamicSharedMemorySize,
                            ATTN_LDS);
  attn_fused_kernel<<<dim3(S_ / 16, B_ * H_), 256, ATTN_LDS, stream>>>(
      qh, kh, vt, mask, attnp, cx);

  constexpr int FC_LDS = 16 * EMB_ * 2 + 16 * EMB_ * 4;
  (void)hipFuncSetAttribute((const void*)fc_ln_kernel,
                            hipFuncAttributeMaxDynamicSharedMemorySize, FC_LDS);
  fc_ln_kernel<<<M_ / 16, 256, FC_LDS, stream>>>(cx, wfb, fc_b, in_q, ln_g,
                                                 ln_b, outp);
}
